// Attention_70746701299737
// MI455X (gfx1250) — matrix operand order repaired and checked by reading, not compiled, not run on hardware
//
#include <hip/hip_runtime.h>

typedef __attribute__((ext_vector_type(16))) __bf16 v16bf;
typedef __attribute__((ext_vector_type(8)))  float  v8f;
typedef __attribute__((ext_vector_type(4)))  unsigned int u32x4;
typedef __attribute__((ext_vector_type(8)))  int          i32x8;
typedef __attribute__((ext_vector_type(4)))  int          i32x4;

#define EDIM   768
#define SEQ    49
#define NHEADS 12
#define DHEAD  64
#define NBATCH 1024
#define MTOT   (NBATCH * SEQ)   // 50176 = 392 * 128
#define KSTEPS (EDIM / 32)      // 24

union FragBF { v16bf v; uint4 q[2]; };
union Pack8  { uint4 q;    unsigned short h[8];  };
union Pack16 { uint4 q[2]; unsigned short h[16]; };
union Pack32 { uint4 q[4]; unsigned short h[32]; };

static __device__ __forceinline__ unsigned short f2bf(float f) {
  __bf16 b = (__bf16)f;
  return __builtin_bit_cast(unsigned short, b);
}

static __device__ __forceinline__ v8f wmma_bf16(v16bf a, v16bf b, v8f c) {
  // D = A(16x32 bf16) * B(32x16 bf16) + C(16x16 f32)
  return __builtin_amdgcn_wmma_f32_16x16x32_bf16(false, a, false, b, (short)0, c,
                                                 false, false);
}

// ---------------------------------------------------------------------------
// TDM: DMA a 2D bf16 tile (tile_d1 rows x tile_d0 elems, row stride
// `stride_el` elements) from global memory into LDS at byte offset lds_off.
// Issued once per wave; completion tracked by TENSORcnt.
// ---------------------------------------------------------------------------
static __device__ __forceinline__ void tdm_load_2d(unsigned lds_off,
                                                   unsigned long long gaddr,
                                                   unsigned tile_d0,
                                                   unsigned tile_d1,
                                                   unsigned stride_el) {
  u32x4 g0;
  g0[0] = 1u;                                             // count=1 (valid D#)
  g0[1] = lds_off;                                        // lds_addr (bytes)
  g0[2] = (unsigned)(gaddr & 0xFFFFFFFFu);                // global_addr[31:0]
  g0[3] = (unsigned)((gaddr >> 32) & 0x01FFFFFFu)         // global_addr[56:32]
          | (2u << 30);                                   // type=2 ("image")
  const unsigned td = 0x40000000u;                        // huge tensor dims (no OOB)
  i32x8 g1;
  g1[0] = (int)(1u << 16);                                // data_size=1 (2B), wg_mask=0
  g1[1] = (int)((td & 0xFFFFu) << 16);                    // tensor_dim0[15:0]
  g1[2] = (int)(((td >> 16) & 0xFFFFu) | ((td & 0xFFFFu) << 16));  // dim0 hi | dim1 lo
  g1[3] = (int)(((td >> 16) & 0xFFFFu) | (tile_d0 << 16));         // dim1 hi | tile_dim0
  g1[4] = (int)(tile_d1 & 0xFFFFu);                       // tile_dim1 | tile_dim2=0
  g1[5] = (int)stride_el;                                 // tensor_dim0_stride[31:0]
  g1[6] = 0;                                              // stride0 hi | stride1 lo
  g1[7] = 0;                                              // stride1 hi
  i32x4 g2; g2[0] = 0; g2[1] = 0; g2[2] = 0; g2[3] = 0;
  i32x4 g3; g3[0] = 0; g3[1] = 0; g3[2] = 0; g3[3] = 0;
  i32x8 g4;
  g4[0] = 0; g4[1] = 0; g4[2] = 0; g4[3] = 0;
  g4[4] = 0; g4[5] = 0; g4[6] = 0; g4[7] = 0;
  __builtin_amdgcn_tensor_load_to_lds(g0, g1, g2, g3, g4, 0);
}

// ---------------------------------------------------------------------------
// fp32 -> bf16 bulk convert (8 elements / thread, 128b loads & stores)
// ---------------------------------------------------------------------------
__global__ __launch_bounds__(256) void k_cvt_f32_bf16(const float* __restrict__ s,
                                                      unsigned short* __restrict__ d,
                                                      int n8) {
  int i = blockIdx.x * 256 + threadIdx.x;
  if (i >= n8) return;
  const float4* sp = (const float4*)s;
  float4 f0 = sp[2 * i], f1 = sp[2 * i + 1];
  Pack8 o;
  o.h[0] = f2bf(f0.x); o.h[1] = f2bf(f0.y); o.h[2] = f2bf(f0.z); o.h[3] = f2bf(f0.w);
  o.h[4] = f2bf(f1.x); o.h[5] = f2bf(f1.y); o.h[6] = f2bf(f1.z); o.h[7] = f2bf(f1.w);
  ((uint4*)d)[i] = o.q;
}

// ---------------------------------------------------------------------------
// fp32 (K,N) -> bf16 transposed (N,K): 32x32 LDS-tiled transpose.
// Removes the per-k-step B transpose scatter from the GEMM hot loop.
// ---------------------------------------------------------------------------
__global__ __launch_bounds__(256) void k_cvt_transpose(const float* __restrict__ s,
                                                       unsigned short* __restrict__ d) {
  __shared__ unsigned short tile[32][33];
  const int bx = blockIdx.x * 32;        // source col (n)
  const int by = blockIdx.y * 32;        // source row (k)
  const int tx = threadIdx.x & 31;
  const int ty = threadIdx.x >> 5;       // 0..7
#pragma unroll
  for (int r = ty; r < 32; r += 8)
    tile[r][tx] = f2bf(s[(size_t)(by + r) * EDIM + bx + tx]);
  __syncthreads();
#pragma unroll
  for (int r = ty; r < 32; r += 8)
    d[(size_t)(bx + r) * EDIM + by + tx] = tile[tx][r];
}

// ---------------------------------------------------------------------------
// Tiled WMMA GEMM: C(M,768) = A(M,768)bf16 * Bt(768,768)^T bf16 + bias
// Bt is pre-transposed (N,K). Block tile 128x128, k-step 32, double-buffered
// LDS filled by the Tensor Data Mover (wave 0 issues, TENSORcnt pipelined).
// 8 waves -> 32x64 per wave (2x4 wmma tiles).
// rope=1: fused 2D-RoPE epilogue (Q/K). store_f32: fp32 out (final proj).
// ---------------------------------------------------------------------------
__global__ __launch_bounds__(256) void k_gemm_bf16(
    const unsigned short* __restrict__ A,
    const unsigned short* __restrict__ Bt,
    const float* __restrict__ bias,
    void* __restrict__ out,
    int rope, int store_f32) {
  __shared__ __align__(16) unsigned short As[2][128 * 32];   // [m][k]
  __shared__ __align__(16) unsigned short Bts[2][128 * 32];  // [n][k]

  const int t    = threadIdx.x;
  const int lane = t & 31;
  const int wave = t >> 5;
  const int lg   = lane >> 4;   // lane group (0/1)
  const int lm   = lane & 15;
  const int wm   = wave & 3;    // m offset 32*wm
  const int wn   = wave >> 2;   // n offset 64*wn
  const int m_blk = blockIdx.x * 128;
  const int n_blk = blockIdx.y * 128;

  const unsigned ldsA0 = (unsigned)(uintptr_t)&As[0][0];
  const unsigned ldsA1 = (unsigned)(uintptr_t)&As[1][0];
  const unsigned ldsB0 = (unsigned)(uintptr_t)&Bts[0][0];
  const unsigned ldsB1 = (unsigned)(uintptr_t)&Bts[1][0];
  const unsigned long long gA = (unsigned long long)(uintptr_t)(A  + (size_t)m_blk * EDIM);
  const unsigned long long gB = (unsigned long long)(uintptr_t)(Bt + (size_t)n_blk * EDIM);

  v8f acc[2][4];
  for (int mi = 0; mi < 2; mi++)
    for (int ni = 0; ni < 4; ni++)
      for (int e = 0; e < 8; e++) acc[mi][ni][e] = 0.0f;

  // prologue: TDM tile 0 -> buffer 0
  if (wave == 0) {
    tdm_load_2d(ldsA0, gA, 32, 128, EDIM);
    tdm_load_2d(ldsB0, gB, 32, 128, EDIM);
  }

  for (int i = 0; i < KSTEPS; i++) {
    const int cur = i & 1;
    if (wave == 0) {
      if (i + 1 < KSTEPS) {  // prefetch tile i+1 into the free buffer
        const unsigned long long koff = (unsigned long long)(i + 1) * 64ull;  // 32 el * 2B
        tdm_load_2d(cur ? ldsA0 : ldsA1, gA + koff, 32, 128, EDIM);
        tdm_load_2d(cur ? ldsB0 : ldsB1, gB + koff, 32, 128, EDIM);
        __builtin_amdgcn_s_wait_tensorcnt(2);  // in-order: tile i has landed
      } else {
        __builtin_amdgcn_s_wait_tensorcnt(0);
      }
    }
    __syncthreads();  // publish tile i to all waves

    FragBF af[2];
#pragma unroll
    for (int mi = 0; mi < 2; mi++) {
      const unsigned short* ap = &As[cur][(wm * 32 + mi * 16 + lm) * 32 + lg * 8];
      af[mi].q[0] = *(const uint4*)ap;          // K = 8*lg + 0..7
      af[mi].q[1] = *(const uint4*)(ap + 16);   // K = 16 + 8*lg + 0..7
    }
#pragma unroll
    for (int ni = 0; ni < 4; ni++) {
      FragBF bf;
      const unsigned short* bp = &Bts[cur][(wn * 64 + ni * 16 + lm) * 32 + lg * 8];
      bf.q[0] = *(const uint4*)bp;              // K = 8*lg + 0..7
      bf.q[1] = *(const uint4*)(bp + 16);       // K = 16 + 8*lg + 0..7
      acc[0][ni] = wmma_bf16(af[0].v, bf.v, acc[0][ni]);
      acc[1][ni] = wmma_bf16(af[1].v, bf.v, acc[1][ni]);
    }
    __syncthreads();  // reads of buffer `cur` done before TDM overwrites it
  }

  // epilogue: bias (+ fused 2D RoPE via lane-pair shuffle), store
#pragma unroll
  for (int mi = 0; mi < 2; mi++)
#pragma unroll
    for (int ni = 0; ni < 4; ni++) {
      const int col = n_blk + wn * 64 + ni * 16 + lm;
      const float bv = bias[col];
#pragma unroll
      for (int j = 0; j < 8; j++) {
        const int row = m_blk + wm * 32 + mi * 16 + lg * 8 + j;
        float val = acc[mi][ni][j] + bv;
        if (rope) {
          const int d = col & 63;                 // head-dim position
          const int s = row % SEQ;                // grid position
          const float pos  = (d & 32) ? (float)(s % 7) : (float)(s / 7);
          // inv_freq = 100^(-(2*jj)/32) = exp(-jj * ln(100)/16)
          const float invf = __expf(-0.28782313f * (float)((d & 31) >> 1));
          const float ang  = pos * invf;
          const float cs = __cosf(ang), sn = __sinf(ang);
          const float pv = __shfl_xor(val, 1, 32);  // partner column d^1
          val = (d & 1) ? (pv * sn + val * cs) : (val * cs - pv * sn);
        }
        const size_t o = (size_t)row * EDIM + col;
        if (store_f32) ((float*)out)[o] = val;
        else           ((unsigned short*)out)[o] = f2bf(val);
      }
    }
}

// ---------------------------------------------------------------------------
// Fused attention per (batch, head): S=QK^T (pad 49->64), softmax, P@V.
// 4 waves, each owns a 16-row stripe. attn_weight -> d_out, ctx -> bf16 ws.
// ---------------------------------------------------------------------------
__global__ __launch_bounds__(128) void k_attention(
    const unsigned short* __restrict__ Q,
    const unsigned short* __restrict__ K,
    const unsigned short* __restrict__ V,
    unsigned short* __restrict__ ctx,
    float* __restrict__ attnw) {
  __shared__ __align__(16) unsigned short Qs[64 * 64];  // [q][d]
  __shared__ __align__(16) unsigned short Ks[64 * 64];  // [k][d]
  __shared__ __align__(16) unsigned short Vt[64 * 64];  // transposed [d][k]
  __shared__ __align__(16) float Ps[64 * 64];           // scores / probs

  const int b = blockIdx.x / NHEADS;
  const int h = blockIdx.x % NHEADS;
  const int t = threadIdx.x;
  const int lane = t & 31, wave = t >> 5;
  const int lg = lane >> 4, lm = lane & 15;

  {  // cooperative load with zero-padding of rows 49..63
    const int row = t >> 1;
    const int cg  = (t & 1) << 5;  // 0 / 32 columns
    if (row < SEQ) {
      const size_t base = ((size_t)(b * SEQ + row)) * EDIM + h * DHEAD + cg;
      const uint4* qp = (const uint4*)(Q + base);
      const uint4* kp = (const uint4*)(K + base);
      uint4* qd = (uint4*)&Qs[row * 64 + cg];
      uint4* kd = (uint4*)&Ks[row * 64 + cg];
#pragma unroll
      for (int i = 0; i < 4; i++) { qd[i] = qp[i]; kd[i] = kp[i]; }
      Pack32 vv;
      const uint4* vp = (const uint4*)(V + base);
#pragma unroll
      for (int i = 0; i < 4; i++) vv.q[i] = vp[i];
#pragma unroll
      for (int c = 0; c < 32; c++) Vt[(cg + c) * 64 + row] = vv.h[c];
    } else {
      uint4 z; z.x = z.y = z.z = z.w = 0u;
      uint4* qd = (uint4*)&Qs[row * 64 + cg];
      uint4* kd = (uint4*)&Ks[row * 64 + cg];
#pragma unroll
      for (int i = 0; i < 4; i++) { qd[i] = z; kd[i] = z; }
#pragma unroll
      for (int c = 0; c < 32; c++) Vt[(cg + c) * 64 + row] = 0;
    }
  }
  __syncthreads();

  const int m0 = wave * 16;

  // Phase 1: S = Q @ K^T (64x64, Dh=64 = 2 k-steps of 32)
  FragBF aq[2];
#pragma unroll
  for (int ks = 0; ks < 2; ks++) {
    const unsigned short* p = &Qs[(m0 + lm) * 64 + ks * 32 + lg * 8];
    aq[ks].q[0] = *(const uint4*)p;
    aq[ks].q[1] = *(const uint4*)(p + 16);
  }
  v8f accS[4];
  for (int nt = 0; nt < 4; nt++)
    for (int e = 0; e < 8; e++) accS[nt][e] = 0.0f;
#pragma unroll
  for (int nt = 0; nt < 4; nt++) {
#pragma unroll
    for (int ks = 0; ks < 2; ks++) {
      FragBF bk;  // B[kdim=d][n=key] = K[key][d]
      const unsigned short* p = &Ks[(nt * 16 + lm) * 64 + ks * 32 + lg * 8];
      bk.q[0] = *(const uint4*)p;
      bk.q[1] = *(const uint4*)(p + 16);
      accS[nt] = wmma_bf16(aq[ks].v, bk.v, accS[nt]);
    }
  }
#pragma unroll
  for (int nt = 0; nt < 4; nt++)
#pragma unroll
    for (int j = 0; j < 8; j++)
      Ps[(m0 + lg * 8 + j) * 64 + nt * 16 + lm] = accS[nt][j];
  __syncthreads();

  // Softmax: one thread per row; scale 1/sqrt(64)=0.125; mask padded keys
  if (t < 64) {
    float* prow = &Ps[t * 64];
    if (t < SEQ) {
      const float scale = 0.125f;
      float mx = -3.0e38f;
      for (int c = 0; c < SEQ; c++) mx = fmaxf(mx, prow[c]);
      float sum = 0.0f;
      for (int c = 0; c < SEQ; c++) {
        float e_ = __expf((prow[c] - mx) * scale);
        sum += e_;
        prow[c] = e_;
      }
      const float inv = 1.0f / sum;
      float* aw = attnw + ((size_t)(b * NHEADS + h) * SEQ + t) * SEQ;
      for (int c = 0; c < SEQ; c++) {
        float p_ = prow[c] * inv;
        prow[c] = p_;
        aw[c] = p_;
      }
      for (int c = SEQ; c < 64; c++) prow[c] = 0.0f;
    } else {
      for (int c = 0; c < 64; c++) prow[c] = 0.0f;
    }
  }
  __syncthreads();

  // Phase 2: ctx = P @ V
  FragBF ap[2];
#pragma unroll
  for (int ks = 0; ks < 2; ks++) {
    const float* p = &Ps[(m0 + lm) * 64 + ks * 32 + lg * 8];
    float tmp[16];
    *(float4*)&tmp[0]  = *(const float4*)(p);
    *(float4*)&tmp[4]  = *(const float4*)(p + 4);
    *(float4*)&tmp[8]  = *(const float4*)(p + 16);
    *(float4*)&tmp[12] = *(const float4*)(p + 20);
    Pack16 pk;
#pragma unroll
    for (int e = 0; e < 16; e++) pk.h[e] = f2bf(tmp[e]);
    ap[ks].q[0] = pk.q[0];
    ap[ks].q[1] = pk.q[1];
  }
  v8f accC[4];
  for (int nt = 0; nt < 4; nt++)
    for (int e = 0; e < 8; e++) accC[nt][e] = 0.0f;
#pragma unroll
  for (int nt = 0; nt < 4; nt++) {
#pragma unroll
    for (int ks = 0; ks < 2; ks++) {
      FragBF bv;  // B[kdim=key][n=d] = V[key][d] = Vt[d][key]
      const unsigned short* p = &Vt[(nt * 16 + lm) * 64 + ks * 32 + lg * 8];
      bv.q[0] = *(const uint4*)p;
      bv.q[1] = *(const uint4*)(p + 16);
      accC[nt] = wmma_bf16(ap[ks].v, bv.v, accC[nt]);
    }
  }
#pragma unroll
  for (int nt = 0; nt < 4; nt++)
#pragma unroll
    for (int j = 0; j < 8; j++) {
      const int r = m0 + lg * 8 + j;
      if (r < SEQ) {
        const int d = nt * 16 + lm;
        ctx[((size_t)(b * SEQ + r)) * EDIM + h * DHEAD + d] = f2bf(accC[nt][j]);
      }
    }
}

// ---------------------------------------------------------------------------
extern "C" void kernel_launch(void* const* d_in, const int* in_sizes, int n_in,
                              void* d_out, int out_size, void* d_ws, size_t ws_size,
                              hipStream_t stream) {
  (void)in_sizes; (void)n_in; (void)out_size; (void)ws_size;
  const float* x  = (const float*)d_in[0];
  const float* Wq = (const float*)d_in[1];
  const float* bq = (const float*)d_in[2];
  const float* Wk = (const float*)d_in[3];
  const float* bk = (const float*)d_in[4];
  const float* Wv = (const float*)d_in[5];
  const float* bv = (const float*)d_in[6];
  const float* Wo = (const float*)d_in[7];
  const float* bo = (const float*)d_in[8];
  float* out = (float*)d_out;

  const size_t xbytes = (size_t)MTOT * EDIM * sizeof(unsigned short);
  const size_t wbytes = (size_t)EDIM * EDIM * sizeof(unsigned short);
  char* p = (char*)d_ws;
  unsigned short* xb  = (unsigned short*)p; p += xbytes;
  unsigned short* Wqt = (unsigned short*)p; p += wbytes;   // transposed bf16
  unsigned short* Wkt = (unsigned short*)p; p += wbytes;
  unsigned short* Wvt = (unsigned short*)p; p += wbytes;
  unsigned short* Wot = (unsigned short*)p; p += wbytes;
  unsigned short* Qb  = (unsigned short*)p; p += xbytes;
  unsigned short* Kb  = (unsigned short*)p; p += xbytes;
  unsigned short* Vb  = (unsigned short*)p; p += xbytes;
  unsigned short* Cb  = (unsigned short*)p; p += xbytes;

  // 1) bf16 staging: x flat, weights transposed (N,K)
  {
    int n8 = (int)((size_t)MTOT * EDIM / 8);
    k_cvt_f32_bf16<<<(n8 + 255) / 256, 256, 0, stream>>>(x, xb, n8);
    dim3 tg(EDIM / 32, EDIM / 32);
    k_cvt_transpose<<<tg, 256, 0, stream>>>(Wq, Wqt);
    k_cvt_transpose<<<tg, 256, 0, stream>>>(Wk, Wkt);
    k_cvt_transpose<<<tg, 256, 0, stream>>>(Wv, Wvt);
    k_cvt_transpose<<<tg, 256, 0, stream>>>(Wo, Wot);
  }

  // 2) QKV projections (RoPE fused into Q/K epilogues)
  dim3 gg(MTOT / 128, EDIM / 128), gb(256);
  k_gemm_bf16<<<gg, gb, 0, stream>>>(xb, Wqt, bq, (void*)Qb, 1, 0);
  k_gemm_bf16<<<gg, gb, 0, stream>>>(xb, Wkt, bk, (void*)Kb, 1, 0);
  k_gemm_bf16<<<gg, gb, 0, stream>>>(xb, Wvt, bv, (void*)Vb, 0, 0);

  // 3) fused attention (writes attn_weight directly to d_out tail)
  float* attnw = out + (size_t)MTOT * EDIM;
  k_attention<<<NBATCH * NHEADS, 128, 0, stream>>>(Qb, Kb, Vb, Cb, attnw);

  // 4) output projection -> fp32 attn_value at d_out head
  k_gemm_bf16<<<gg, gb, 0, stream>>>(Cb, Wot, bo, (void*)out, 0, 1);
}
